// CustomLSTM_1932735283464
// MI455X (gfx1250) — matrix operand order repaired and checked by reading, not compiled, not run on hardware
//
#include <hip/hip_runtime.h>

typedef __attribute__((ext_vector_type(16))) _Float16 v16h;
typedef __attribute__((ext_vector_type(8)))  _Float16 v8h;
typedef __attribute__((ext_vector_type(8)))  float    v8f;

#define B_    256
#define T_    512
#define IN_   8
#define H_    512
#define G4    2048            // 4*H
#define HS    16              // hidden units per workgroup
#define NSLICE (H_/HS)        // 32 slices per batch tile
#define KH    512             // K extent of h part
#define KX0   32              // padded x K extent, layer 0
#define KC0   ((KH+KX0)/32)   // 17 k-chunks, layer 0
#define KC1   ((KH+H_)/32)    // 32 k-chunks, layer 1
#define ASTRIDE (1024+8)      // LDS A row stride in halves (bank-conflict pad)
#define GSTRIDE (16+4)        // LDS gatesT row stride in floats

// Dynamic LDS bytes for a layer: W block + A stage + gatesT + cell state
#define LDS_W(KC)   ((size_t)(KC) * 4096)              // 4 tiles * KC * 512 halves
#define LDS_A       ((size_t)16 * ASTRIDE * 2)         // 33024
#define LDS_G       ((size_t)64 * GSTRIDE * 4)         // 5120
#define LDS_C       ((size_t)16 * HS * 4)              // 1024
#define LDS_TOTAL(KC) (LDS_W(KC) + LDS_A + LDS_G + LDS_C)

// ---------------------------------------------------------------------------
// Weight packing: combined [W_hh | W_ih] -> WMMA B-fragment order, with
// gate-interleaved columns (packed col pc = 4*n + g, PyTorch gate order ifgo).
// Flat layout: ((nt*KC + kc)*32 + lane)*16 + j ; element (K,N):
//   N = nt*16 + lane%16 , K = kc*32 + (lane/16)*16 + j
// Fragments for one 64-col slice (4 tiles) are contiguous -> straight LDS copy.
// A operand layout per step: K[0..511] = h_{t-1}, K[512..512+KX) = x_t.
// ---------------------------------------------------------------------------
__global__ void pack_w_kernel(const float* __restrict__ w_ih,
                              const float* __restrict__ w_hh,
                              _Float16* __restrict__ dst, int KXin, int KC)
{
    size_t total = (size_t)128 * KC * 512;
    for (size_t id = (size_t)blockIdx.x * 256 + threadIdx.x; id < total;
         id += (size_t)gridDim.x * 256) {
        int j    = (int)(id & 15);
        int lane = (int)((id >> 4) & 31);
        size_t fi = id >> 9;               // nt*KC + kc
        int kc  = (int)(fi % KC);
        int ntl = (int)(fi / KC);
        int pc  = ntl * 16 + (lane & 15);  // packed gate-column
        int n   = pc >> 2, g = pc & 3;
        int row = g * H_ + n;              // original weight row
        int k   = kc * 32 + (lane >> 4) * 16 + j;
        float val = 0.f;
        if (k < KH) {
            val = w_hh[(size_t)row * H_ + k];
        } else {
            int ki = k - KH;
            if (ki < KXin) val = w_ih[(size_t)row * KXin + ki];
        }
        dst[id] = (_Float16)val;
    }
}

__global__ void pack_bias_kernel(const float* __restrict__ bi,
                                 const float* __restrict__ bh,
                                 float* __restrict__ dst)
{
    int pc = blockIdx.x * 256 + threadIdx.x;
    if (pc < G4) {
        int n = pc >> 2, g = pc & 3, row = g * H_ + n;
        dst[pc] = bi[row] + bh[row];
    }
}

__global__ void zero_kernel(unsigned int* __restrict__ p, int n)
{
    int i = blockIdx.x * 256 + threadIdx.x;
    if (i < n) p[i] = 0u;
}

// ---------------------------------------------------------------------------
// Persistent LSTM layer kernel, weights LDS-stationary.
// grid = 16 batch-tiles x 32 hidden-slices = 512 WGs, 128 threads (4 waves).
// Per step: stage [h_{t-1} | x_t] into LDS, each wave runs one 16x16 gate
// tile via v_wmma_f32_16x16x32_f16 (software-pipelined LDS operand fetch),
// then slice-local gate math + cross-slice handoff of h using
// GLOBAL_LOAD_MONITOR_B32 + S_MONITOR_SLEEP (MWAIT) on the step flag.
// ---------------------------------------------------------------------------
__global__ __launch_bounds__(128)
void lstm_kernel(const float*    __restrict__ u,     // [B,T,IN]  (layer 0 only)
                 const _Float16* __restrict__ xh,    // [T,B,H]   (layer 1 only)
                 const _Float16* __restrict__ pw,    // packed weights
                 const float*    __restrict__ biasp, // packed bias [2048]
                 _Float16*       __restrict__ hhist, // layer 0: out0 [T,B,H]
                 _Float16*       __restrict__ hbuf,  // layer 1: [2,B,H]
                 unsigned int*   __restrict__ flags, // [16,T]
                 int KX, int KC)
{
    extern __shared__ char smem[];
    _Float16* sW = reinterpret_cast<_Float16*>(smem);
    _Float16* sA = reinterpret_cast<_Float16*>(smem + LDS_W(KC));
    float*    sG = reinterpret_cast<float*>(smem + LDS_W(KC) + LDS_A);
    float*    sC = reinterpret_cast<float*>(smem + LDS_W(KC) + LDS_A + LDS_G);

    const int tile    = blockIdx.x / NSLICE;
    const int slice   = blockIdx.x % NSLICE;
    const int tid     = threadIdx.x;
    const int lane    = tid & 31;
    const int wv      = tid >> 5;           // wave 0..3
    const int rowbase = tile * 16;

    // ---- one-time: weight block -> LDS (fragments are contiguous) ----
    {
        const float4* wsrc = reinterpret_cast<const float4*>(pw + (size_t)slice * 4 * KC * 512);
        float4* wdst = reinterpret_cast<float4*>(sW);
        const int nvec = KC * 256;          // KC*4096 bytes / 16
        for (int i = tid; i < nvec; i += 128) wdst[i] = wsrc[i];
    }
    for (int e = tid; e < 16 * HS; e += 128) sC[e] = 0.f;
    __syncthreads();

    const float bias = biasp[slice * 64 + wv * 16 + (lane & 15)];
    const _Float16* arow = &sA[(lane & 15) * ASTRIDE + (lane >> 4) * 16];
    const v16h* wfrag = reinterpret_cast<const v16h*>(sW) + (size_t)wv * KC * 32 + lane;

    for (int t = 0; t < T_; ++t) {
        // ---- wait for all slices of this tile to publish h_{t-1} ----
        if (t > 0) {
            if (tid == 0) {
                const unsigned int* fp = &flags[tile * T_ + t - 1];
                for (;;) {
                    unsigned int v;
                    // Arm a GL2 monitor on the flag line, then sleep until the
                    // producer's release-atomic touches it (or timeout).
                    asm volatile("global_load_monitor_b32 %0, %1, off scope:SCOPE_DEV\n\t"
                                 "s_wait_loadcnt 0x0"
                                 : "=v"(v) : "v"(fp) : "memory");
                    if (v >= (unsigned)NSLICE) break;
                    asm volatile("s_monitor_sleep 64" ::: "memory");
                }
                // Confirm with an acquire load to order subsequent h reads.
                (void)__hip_atomic_load(fp, __ATOMIC_ACQUIRE, __HIP_MEMORY_SCOPE_AGENT);
            }
            __syncthreads();
        }
        // ---- stage A: h part, K[0..511] ----
        const _Float16* hsrc = nullptr;
        if (t > 0)
            hsrc = hbuf ? (hbuf + ((size_t)((t - 1) & 1) * B_ + rowbase) * H_)
                        : (hhist + ((size_t)(t - 1) * B_ + rowbase) * H_);
        for (int idx = tid; idx < 16 * (KH / 8); idx += 128) {
            int r = idx >> 6, c8 = idx & 63;
            float4* d = reinterpret_cast<float4*>(&sA[r * ASTRIDE + c8 * 8]);
            if (hsrc) *d = reinterpret_cast<const float4*>(hsrc + (size_t)r * H_)[c8];
            else      *d = make_float4(0.f, 0.f, 0.f, 0.f);
        }
        // ---- stage A: x part, K[512..512+KX) ----
        if (KX == KX0) {
            for (int idx = tid; idx < 16 * KX0; idx += 128) {
                int r = idx >> 5, c = idx & 31;
                sA[r * ASTRIDE + KH + c] =
                    (c < IN_) ? (_Float16)u[((size_t)(rowbase + r) * T_ + t) * IN_ + c]
                              : (_Float16)0.f;
            }
            if (t + 1 < T_ && tid < 16)     // prefetch next step's u rows
                __builtin_prefetch((const void*)(u + ((size_t)(rowbase + tid) * T_ + t + 1) * IN_), 0, 1);
        } else {
            const _Float16* xs = xh + ((size_t)t * B_ + rowbase) * H_;
            for (int idx = tid; idx < 16 * (H_ / 8); idx += 128) {
                int r = idx >> 6, c8 = idx & 63;
                *reinterpret_cast<float4*>(&sA[r * ASTRIDE + KH + c8 * 8]) =
                    reinterpret_cast<const float4*>(xs + (size_t)r * H_)[c8];
            }
            if (t + 1 < T_ && tid < 64)     // prefetch next step's x slab (16 KB)
                __builtin_prefetch((const void*)(xh + ((size_t)(t + 1) * B_ + rowbase) * H_ + (size_t)tid * 128), 0, 1);
        }
        __syncthreads();

        // ---- WMMA: 16x16 tile, operands from LDS, depth-1 software pipeline ----
        v8f acc;
        #pragma unroll
        for (int v = 0; v < 8; ++v) acc[v] = bias;
        {
            auto loadA = [&](int kc) -> v16h {
                v8h a0 = *reinterpret_cast<const v8h*>(arow + kc * 32);
                v8h a1 = *reinterpret_cast<const v8h*>(arow + kc * 32 + 8);
                return __builtin_shufflevector(a0, a1, 0, 1, 2, 3, 4, 5, 6, 7,
                                               8, 9, 10, 11, 12, 13, 14, 15);
            };
            v16h afc = loadA(0);
            v16h bfc = wfrag[0];
            for (int kc = 0; kc + 1 < KC; ++kc) {
                v16h afn = loadA(kc + 1);             // issue next-chunk loads
                v16h bfn = wfrag[(size_t)(kc + 1) * 32];
                acc = __builtin_amdgcn_wmma_f32_16x16x32_f16(
                    false, afc, false, bfc, (short)0, acc, false, false);
                afc = afn; bfc = bfn;
            }
            acc = __builtin_amdgcn_wmma_f32_16x16x32_f16(
                false, afc, false, bfc, (short)0, acc, false, false);
        }
        // ---- stage accumulators to LDS (gatesT[col][row]) ----
        {
            float* gp = &sG[(wv * 16 + (lane & 15)) * GSTRIDE + (lane >> 4) * 8];
            *reinterpret_cast<float4*>(gp)     = make_float4(acc[0], acc[1], acc[2], acc[3]);
            *reinterpret_cast<float4*>(gp + 4) = make_float4(acc[4], acc[5], acc[6], acc[7]);
        }
        __syncthreads();

        // ---- element-wise gate math + state update for this slice ----
        for (int e = tid; e < 16 * HS; e += 128) {
            int r = e & 15, nh = e >> 4;
            float gi = sG[(nh * 4 + 0) * GSTRIDE + r];
            float gf = sG[(nh * 4 + 1) * GSTRIDE + r];
            float gg = sG[(nh * 4 + 2) * GSTRIDE + r];
            float go = sG[(nh * 4 + 3) * GSTRIDE + r];
            float i_ = 1.f / (1.f + __expf(-gi));
            float f_ = 1.f / (1.f + __expf(-gf));
            float g_ = tanhf(gg);
            float o_ = 1.f / (1.f + __expf(-go));
            float c  = f_ * sC[e] + i_ * g_;
            sC[e] = c;
            _Float16 hv = (_Float16)(o_ * tanhf(c));
            int n = slice * HS + nh;
            if (hhist) hhist[((size_t)t * B_ + rowbase + r) * H_ + n] = hv;
            if (hbuf)  hbuf[((size_t)(t & 1) * B_ + rowbase + r) * H_ + n] = hv;
        }
        __threadfence();
        __syncthreads();
        if (tid == 0)
            __hip_atomic_fetch_add(&flags[tile * T_ + t], 1u,
                                   __ATOMIC_RELEASE, __HIP_MEMORY_SCOPE_AGENT);
    }
}

// ---------------------------------------------------------------------------
// FC head: out = ((tanh(hT @ W0^T + b0) @ W1^T + b1) + 1)/2 * (4.2-2.5) + 2.5
// ---------------------------------------------------------------------------
__global__ __launch_bounds__(256)
void fc_kernel(const _Float16* __restrict__ hT,   // [B,H]
               const float* __restrict__ w0, const float* __restrict__ b0,
               const float* __restrict__ w1, const float* __restrict__ b1,
               float* __restrict__ out)
{
    __shared__ float sh[256];
    int b = blockIdx.x, j = threadIdx.x;
    float acc = b0[j];
    const _Float16* h = hT + (size_t)b * H_;
    const float*    w = w0 + (size_t)j * H_;
    for (int k = 0; k < H_; k += 4)
        acc += (float)h[k] * w[k] + (float)h[k + 1] * w[k + 1]
             + (float)h[k + 2] * w[k + 2] + (float)h[k + 3] * w[k + 3];
    sh[j] = tanhf(acc);
    __syncthreads();
    if (j < 2) {
        float a = b1[j];
        for (int q = 0; q < 256; ++q) a += sh[q] * w1[j * 256 + q];
        out[b * 2 + j] = (a + 1.f) * 0.5f * (4.2f - 2.5f) + 2.5f;
    }
}

// ---------------------------------------------------------------------------
extern "C" void kernel_launch(void* const* d_in, const int* in_sizes, int n_in,
                              void* d_out, int out_size, void* d_ws, size_t ws_size,
                              hipStream_t stream)
{
    (void)in_sizes; (void)n_in; (void)out_size; (void)ws_size;
    const float* u    = (const float*)d_in[0];
    const float* wih0 = (const float*)d_in[1];
    const float* whh0 = (const float*)d_in[2];
    const float* bih0 = (const float*)d_in[3];
    const float* bhh0 = (const float*)d_in[4];
    const float* wih1 = (const float*)d_in[5];
    const float* whh1 = (const float*)d_in[6];
    const float* bih1 = (const float*)d_in[7];
    const float* bhh1 = (const float*)d_in[8];
    const float* fw0  = (const float*)d_in[9];
    const float* fb0  = (const float*)d_in[10];
    const float* fw1  = (const float*)d_in[11];
    const float* fb1  = (const float*)d_in[12];

    char* ws = (char*)d_ws;
    auto au = [](size_t x) { return (x + 255) & ~(size_t)255; };
    size_t off = 0;
    _Float16* pw0 = (_Float16*)(ws + off); off = au(off + (size_t)128 * KC0 * 512 * 2);
    _Float16* pw1 = (_Float16*)(ws + off); off = au(off + (size_t)128 * KC1 * 512 * 2);
    float*    bp0 = (float*)(ws + off);    off = au(off + (size_t)G4 * 4);
    float*    bp1 = (float*)(ws + off);    off = au(off + (size_t)G4 * 4);
    _Float16* out0 = (_Float16*)(ws + off); off = au(off + (size_t)T_ * B_ * H_ * 2);
    _Float16* hbuf = (_Float16*)(ws + off); off = au(off + (size_t)2 * B_ * H_ * 2);
    unsigned int* flg = (unsigned int*)(ws + off); off = au(off + (size_t)2 * 16 * T_ * 4);
    unsigned int* flg0 = flg;
    unsigned int* flg1 = flg + 16 * T_;

    // Opt in to >64KB dynamic LDS (host-side attribute, not a stream op).
    (void)hipFuncSetAttribute(reinterpret_cast<const void*>(lstm_kernel),
                              hipFuncAttributeMaxDynamicSharedMemorySize,
                              (int)LDS_TOTAL(KC1));

    zero_kernel<<<(2 * 16 * T_ + 255) / 256, 256, 0, stream>>>(flg, 2 * 16 * T_);
    pack_w_kernel<<<1024, 256, 0, stream>>>(wih0, whh0, pw0, IN_, KC0);
    pack_w_kernel<<<2048, 256, 0, stream>>>(wih1, whh1, pw1, H_, KC1);
    pack_bias_kernel<<<8, 256, 0, stream>>>(bih0, bhh0, bp0);
    pack_bias_kernel<<<8, 256, 0, stream>>>(bih1, bhh1, bp1);

    lstm_kernel<<<16 * NSLICE, 128, (unsigned)LDS_TOTAL(KC0), stream>>>(
        u, nullptr, pw0, bp0, out0, nullptr, flg0, KX0, KC0);
    lstm_kernel<<<16 * NSLICE, 128, (unsigned)LDS_TOTAL(KC1), stream>>>(
        nullptr, out0, pw1, bp1, nullptr, hbuf, flg1, H_, KC1);

    const _Float16* hT = hbuf + (size_t)((T_ - 1) & 1) * B_ * H_;
    fc_kernel<<<B_, 256, 0, stream>>>(hT, fw0, fb0, fw1, fb1, (float*)d_out);
}
